// Generator_13142599925896
// MI455X (gfx1250) — matrix operand order repaired and checked by reading, not compiled, not run on hardware
//
#include <hip/hip_runtime.h>
#include <hip/hip_bf16.h>

// ---------------------------------------------------------------------------
// MI455X (gfx1250) implementation of the DGCNN-style generator.
// All dense contractions go through V_WMMA_F32_16X16X32_F16 (wave32 WMMA).
// Uniform data layout: point-major, X[point][channel] (channels contiguous),
// so every conv is a GEMM  out[p][o] = sum_k W[o][k] * in[p][k].
// WMMA fragments are loaded straight from global memory (per-lane contiguous
// runs per ISA 7.12.2) -- no LDS staging, no barriers, uniform control flow.
// ---------------------------------------------------------------------------

typedef __attribute__((ext_vector_type(16))) _Float16 v16h;
typedef __attribute__((ext_vector_type(8)))  float    v8f;

#define PTS   16384L          // B*N = 8*2048
#define P20   327680L         // B*N*K
#define NPT   2048
#define NB    8

// ----------------- fragment loaders (branch-free, clamped tail) ------------
// A (16x32 f16): lane(half=lane/16, mrow=lane%16) holds row m0+mrow,
//   a[h<8]  = W[row][kc + half*8 + h]
//   a[h>=8] = W[row][kc + 16 + half*8 + (h-8)]
template <bool TAIL>
__device__ __forceinline__ v16h load_afrag(const float* __restrict__ W,
                                           long row, int K, int kc, int half)
{
    const float* ar = W + (size_t)row * K;
    v16h a;
    #pragma unroll
    for (int h = 0; h < 16; ++h) {
        int ka = (h < 8) ? (kc + half * 8 + h) : (kc + 16 + half * 8 + (h - 8));
        int kcl = TAIL ? ((ka < K) ? ka : (K - 1)) : ka;
        float v = ar[kcl];
        if (TAIL && ka >= K) v = 0.f;
        a[h] = (_Float16)v;
    }
    return a;
}

// B (32x16 f16): lane holds column p = p0 + (lane%16),
//   b[h<8]  = X[p*ldb + kc + half*8 + h]
//   b[h>=8] = X[p*ldb + kc + 16 + half*8 + (h-8)]
template <bool TAIL>
__device__ __forceinline__ v16h load_bfrag(const float* __restrict__ X,
                                           long p, long ldb, int K, int kc, int half)
{
    const float* col = X + p * ldb;
    v16h b;
    #pragma unroll
    for (int h = 0; h < 16; ++h) {
        int kk = (h < 8) ? (kc + half * 8 + h) : (kc + 16 + half * 8 + (h - 8));
        int kcl = TAIL ? ((kk < K) ? kk : (K - 1)) : kk;
        float v = col[kcl];
        if (TAIL && kk >= K) v = 0.f;
        b[h] = (_Float16)v;
    }
    return b;
}

// ------------------------- generic WMMA GEMM -------------------------------
// W  : (M x K) f32 row-major weights (M is always a multiple of 16 here)
// X  : point-major input, column p of the GEMM is X[p*ldb .. p*ldb+K-1]
// Y  : (Np x ldo) f32, Y[p*ldo + m]
// act: 0 = none, 1 = leaky-relu(slope)
__global__ __launch_bounds__(256)
void gemm_pk(const float* __restrict__ W, const float* __restrict__ X,
             const float* __restrict__ bias, float* __restrict__ Y,
             int M, int K, long Np, long ldb, long ldo, float slope, int act)
{
    const int tid  = threadIdx.x;
    const int wave = tid >> 5;
    const int lane = tid & 31;
    const int half = (lane < 16) ? 0 : 1;
    const int mrow = lane & 15;
    const int m0   = blockIdx.y * 16;
    const long p0  = (long)blockIdx.x * 128 + (long)wave * 16;

    const long arow = m0 + mrow;                 // always < M (M % 16 == 0)
    long pcol = p0 + mrow;                       // clamp for tiny-Np tiles
    if (pcol > Np - 1) pcol = Np - 1;            // loads in-bounds; stores masked

    v8f acc = {0.f, 0.f, 0.f, 0.f, 0.f, 0.f, 0.f, 0.f};

    int kc = 0;
    for (; kc + 32 <= K; kc += 32) {
        v16h a = load_afrag<false>(W, arow, K, kc, half);
        v16h b = load_bfrag<false>(X, pcol, ldb, K, kc, half);
        acc = __builtin_amdgcn_wmma_f32_16x16x32_f16(
                  false, a, false, b, (short)0, acc, false, false);
    }
    if (kc < K) {
        v16h a = load_afrag<true>(W, arow, K, kc, half);
        v16h b = load_bfrag<true>(X, pcol, ldb, K, kc, half);
        acc = __builtin_amdgcn_wmma_f32_16x16x32_f16(
                  false, a, false, b, (short)0, acc, false, false);
    }

    // C/D layout: VGPR v -> M = m0 + v + (lane<16 ? 0 : 8), N = lane&15
    const long p = p0 + mrow;
    if (p < Np) {
        #pragma unroll
        for (int vv = 0; vv < 8; ++vv) {
            int m = m0 + vv + ((lane < 16) ? 0 : 8);
            float r = acc[vv];
            if (bias) r += bias[m];
            if (act == 1) r = (r >= 0.f) ? r : slope * r;
            Y[p * ldo + m] = r;
        }
    }
}

// ------------------------- small helper kernels ----------------------------
__global__ void concat_xz_k(const float* __restrict__ x, const float* __restrict__ z,
                            float* __restrict__ s)
{
    long t = (long)blockIdx.x * blockDim.x + threadIdx.x;
    if (t >= PTS * 131) return;
    long p = t / 131; int c = (int)(t % 131);
    s[t] = (c < 3) ? x[p * 3 + c] : z[p * 128 + (c - 3)];
}

template <int C>
__global__ void knn_k(const float* __restrict__ X, int* __restrict__ idx)
{
    int p = blockIdx.x * blockDim.x + threadIdx.x;
    if (p >= (int)PTS) return;
    int b = p >> 11, n = p & 2047;
    const float* base = X + (size_t)b * NPT * C;
    float xv[C];
    #pragma unroll
    for (int c = 0; c < C; ++c) xv[c] = base[(size_t)n * C + c];
    float bd[20]; int bi[20];
    #pragma unroll
    for (int j = 0; j < 20; ++j) { bd[j] = 3.4e38f; bi[j] = 0; }
    for (int m = 0; m < NPT; ++m) {
        if (m == n) continue;
        const float* xm = base + (size_t)m * C;
        float d = 0.f;
        #pragma unroll
        for (int c = 0; c < C; ++c) { float dv = xv[c] - xm[c]; d += dv * dv; }
        if (d < bd[19]) {
            int j = 19;
            while (j > 0 && bd[j - 1] > d) { bd[j] = bd[j - 1]; bi[j] = bi[j - 1]; --j; }
            bd[j] = d; bi[j] = m;
        }
    }
    for (int j = 0; j < 20; ++j) idx[(size_t)p * 20 + j] = bi[j];
}

template <int C>
__global__ void edge_k(const float* __restrict__ X, const int* __restrict__ idx,
                       float* __restrict__ E)
{
    long t = (long)blockIdx.x * blockDim.x + threadIdx.x;
    if (t >= P20) return;
    long p = t / 20;
    int b = (int)(p >> 11);
    int nb = idx[t];
    const float* xp = X + p * C;
    const float* xn = X + ((size_t)b * NPT + nb) * C;
    float* e = E + t * 2 * C;
    #pragma unroll
    for (int c = 0; c < C; ++c) { float a = xp[c]; e[c] = a; e[C + c] = xn[c] - a; }
}

__global__ void bnstat_k(const float* __restrict__ X, long R, int C,
                         float* __restrict__ mean, float* __restrict__ rstd)
{
    __shared__ float s1[256], s2[256];
    int c = blockIdx.x, tid = threadIdx.x;
    float a = 0.f, q = 0.f;
    for (long r = tid; r < R; r += 256) {
        float v = X[r * C + c]; a += v; q += v * v;
    }
    s1[tid] = a; s2[tid] = q; __syncthreads();
    for (int s = 128; s > 0; s >>= 1) {
        if (tid < s) { s1[tid] += s1[tid + s]; s2[tid] += s2[tid + s]; }
        __syncthreads();
    }
    if (tid == 0) {
        float m = s1[0] / (float)R;
        float v = s2[0] / (float)R - m * m;
        mean[c] = m;
        rstd[c] = rsqrtf(fmaxf(v, 0.f) + 1e-5f);
    }
}

__global__ void bnapply_k(float* __restrict__ X, long R, int C,
                          const float* __restrict__ g, const float* __restrict__ bb,
                          const float* __restrict__ mean, const float* __restrict__ rstd,
                          float slope)
{
    long t = (long)blockIdx.x * blockDim.x + threadIdx.x;
    if (t >= R * C) return;
    int c = (int)(t % C);
    float v = (X[t] - mean[c]) * rstd[c] * g[c] + bb[c];
    X[t] = (v >= 0.f) ? v : slope * v;
}

__global__ void softmax_k(float* __restrict__ W, long P, int C)
{
    long t = (long)blockIdx.x * blockDim.x + threadIdx.x;
    if (t >= P * C) return;
    long p = t / C; int c = (int)(t % C);
    float* base = W + p * 20 * (long)C + c;
    float mx = -3.4e38f;
    #pragma unroll
    for (int j = 0; j < 20; ++j) mx = fmaxf(mx, base[(long)j * C]);
    float e[20]; float s = 0.f;
    #pragma unroll
    for (int j = 0; j < 20; ++j) { e[j] = expf(base[(long)j * C] - mx); s += e[j]; }
    float inv = 1.f / s;
    #pragma unroll
    for (int j = 0; j < 20; ++j) base[(long)j * C] = e[j] * inv;
}

__global__ void mul_k(float* __restrict__ a, const float* __restrict__ b, long n)
{
    long t = (long)blockIdx.x * blockDim.x + threadIdx.x;
    if (t < n) a[t] *= b[t];
}

// repack o_W (F, F, 20) -> (F, 20*F):  Wo[o][j*F + i] = oW[o][i][j]
__global__ void repack_k(const float* __restrict__ oW, float* __restrict__ Wo, int F)
{
    long t = (long)blockIdx.x * blockDim.x + threadIdx.x;
    long tot = (long)F * F * 20;
    if (t >= tot) return;
    long o = t / ((long)F * 20);
    long r = t % ((long)F * 20);
    int i = (int)(r / 20), j = (int)(r % 20);
    Wo[o * (long)F * 20 + (long)j * F + i] = oW[t];
}

__global__ void instat_k(const float* __restrict__ X, int C,
                         float* __restrict__ mean, float* __restrict__ rstd)
{
    __shared__ float s1[256], s2[256];
    int bc = blockIdx.x, tid = threadIdx.x;
    int b = bc / C, c = bc % C;
    float a = 0.f, q = 0.f;
    for (int nn = tid; nn < NPT; nn += 256) {
        float v = X[((size_t)b * NPT + nn) * C + c]; a += v; q += v * v;
    }
    s1[tid] = a; s2[tid] = q; __syncthreads();
    for (int s = 128; s > 0; s >>= 1) {
        if (tid < s) { s1[tid] += s1[tid + s]; s2[tid] += s2[tid + s]; }
        __syncthreads();
    }
    if (tid == 0) {
        float m = s1[0] / (float)NPT;
        float v = s2[0] / (float)NPT - m * m;
        mean[bc] = m;
        rstd[bc] = rsqrtf(fmaxf(v, 0.f) + 1e-5f);
    }
}

__global__ void adain_k(float* __restrict__ X, const float* __restrict__ S,
                        const float* __restrict__ mean, const float* __restrict__ rstd,
                        int C)
{
    long t = (long)blockIdx.x * blockDim.x + threadIdx.x;
    if (t >= PTS * C) return;
    long p = t / C; int c = (int)(t % C);
    int b = (int)(p >> 11);
    float gm = S[p * 2 * C + c];
    float be = S[p * 2 * C + C + c];
    X[t] = gm * (X[t] - mean[b * C + c]) * rstd[b * C + c] + be;
}

__global__ void colmax_k(const float* __restrict__ X, float* __restrict__ fg)
{
    int t = blockIdx.x * blockDim.x + threadIdx.x;
    if (t >= NB * 128) return;
    int b = t / 128, c = t % 128;
    float m = -3.4e38f;
    for (int nn = 0; nn < NPT; ++nn)
        m = fmaxf(m, X[((size_t)b * NPT + nn) * 128 + c]);
    fg[t] = m;
}

__global__ void bn1_k(float* __restrict__ X, int C, const float* __restrict__ g,
                      const float* __restrict__ bb, float slope)
{
    int c = blockIdx.x * blockDim.x + threadIdx.x;
    if (c >= C) return;
    float s = 0.f, q = 0.f;
    for (int b = 0; b < NB; ++b) { float v = X[b * C + c]; s += v; q += v * v; }
    float m = s / (float)NB;
    float var = q / (float)NB - m * m;
    float r = rsqrtf(fmaxf(var, 0.f) + 1e-5f);
    for (int b = 0; b < NB; ++b) {
        float v = (X[b * C + c] - m) * r * g[c] + bb[c];
        X[b * C + c] = (v >= 0.f) ? v : slope * v;
    }
}

__global__ void fc_k(const float* __restrict__ fg, const float* __restrict__ X2,
                     float* __restrict__ FC)
{
    long t = (long)blockIdx.x * blockDim.x + threadIdx.x;
    if (t >= PTS * 640) return;
    long p = t / 640; int c = (int)(t % 640);
    int b = (int)(p >> 11);
    FC[t] = (c < 512) ? fg[b * 512 + c] : X2[p * 128 + (c - 512)];
}

__global__ void final_k(const float* __restrict__ T2, const float* __restrict__ W,
                        const float* __restrict__ bvec, float* __restrict__ out)
{
    int p = blockIdx.x * blockDim.x + threadIdx.x;
    if (p >= (int)PTS) return;
    int b = p >> 11, n = p & 2047;
    float v[64];
    #pragma unroll
    for (int k = 0; k < 64; ++k) v[k] = T2[(size_t)p * 64 + k];
    #pragma unroll
    for (int o = 0; o < 3; ++o) {
        float s = bvec[o];
        #pragma unroll
        for (int k = 0; k < 64; ++k) s += W[o * 64 + k] * v[k];
        out[((size_t)b * 3 + o) * NPT + n] = tanhf(s);
    }
}

// ------------------------- host orchestration ------------------------------
static inline unsigned cdiv(long a, long b) { return (unsigned)((a + b - 1) / b); }

static void gemm(hipStream_t s, const float* W, const float* X, const float* bias,
                 float* Y, int M, int K, long Np, long ldb, long ldo,
                 float slope, int act)
{
    dim3 g(cdiv(Np, 128), (unsigned)((M + 15) / 16));
    gemm_pk<<<g, 256, 0, s>>>(W, X, bias, Y, M, K, Np, ldb, ldo, slope, act);
}

extern "C" void kernel_launch(void* const* d_in, const int* in_sizes, int n_in,
                              void* d_out, int out_size, void* d_ws, size_t ws_size,
                              hipStream_t stream)
{
    (void)in_sizes; (void)n_in; (void)out_size; (void)ws_size;
#define IN(i) ((const float*)d_in[i])
    const float* x = IN(0);
    const float* z = IN(1);

    // ---- workspace arena (floats) ----
    float* wsf = (float*)d_ws;
    size_t off = 0;
    auto alloc = [&](size_t n) { float* p = wsf + off; off += (n + 63) & ~(size_t)63; return p; };
    float* style = alloc(PTS * 128);     // style features, long-lived
    float* sbuf  = alloc(PTS * 256);     // adain gamma/beta (max 2C=256)
    float* x1a   = alloc(PTS * 64);      // block1 output (post adain)
    float* x2a   = alloc(PTS * 128);     // block2 output (post adain)
    int*   idx   = (int*)alloc(P20);     // kNN indices (reused both blocks)
    float* ebuf  = alloc(P20 * 128);     // edge features (max 2C=128)
    float* wA    = alloc(P20 * 64);      // w1 branch
    float* wB    = alloc(P20 * 128);     // w2 branch / softmax weights
    float* xx    = alloc(P20 * 128);     // x branch
    float* fc    = alloc(PTS * 640);     // decoder concat (also style input P*131)
    float* t1o   = alloc(PTS * 256);     // t1 out (also h1 hidden P*128)
    float* t2o   = alloc(PTS * 64);      // t2 out
    float* oWr   = alloc(128 * 2560);    // repacked o_W
    float* mean  = alloc(512);
    float* rstd  = alloc(512);
    float* im    = alloc(NB * 128);
    float* ir    = alloc(NB * 128);
    float* fg    = alloc(NB * 128);
    float* fg1   = alloc(NB * 128);
    float* fg2   = alloc(NB * 512);

    // ================= Phase A: style MLP =================
    concat_xz_k<<<cdiv(PTS * 131, 256), 256, 0, stream>>>(x, z, fc);
    gemm(stream, IN(2), fc, IN(3), t1o, 128, 131, PTS, 131, 128, 0.01f, 1);   // h1
    gemm(stream, IN(4), t1o, IN(5), style, 128, 128, PTS, 128, 128, 0.01f, 1); // h2

    // ================= Phase B: edge block 1 (C=3, fout=64) =================
    knn_k<3><<<cdiv(PTS, 256), 256, 0, stream>>>(x, idx);
    edge_k<3><<<cdiv(P20, 256), 256, 0, stream>>>(x, idx, ebuf);              // (P20,6)
    gemm(stream, IN(6), ebuf + 3, IN(7), wA, 32, 3, P20, 6, 32, 0.f, 0);      // w1
    bnstat_k<<<32, 256, 0, stream>>>(wA, P20, 32, mean, rstd);
    bnapply_k<<<cdiv(P20 * 32, 256), 256, 0, stream>>>(wA, P20, 32, IN(8), IN(9), mean, rstd, 0.01f);
    gemm(stream, IN(10), wA, IN(11), wB, 64, 32, P20, 32, 64, 0.f, 0);        // w2
    bnstat_k<<<64, 256, 0, stream>>>(wB, P20, 64, mean, rstd);
    bnapply_k<<<cdiv(P20 * 64, 256), 256, 0, stream>>>(wB, P20, 64, IN(12), IN(13), mean, rstd, 0.01f);
    softmax_k<<<cdiv(PTS * 64, 256), 256, 0, stream>>>(wB, PTS, 64);
    gemm(stream, IN(14), ebuf, IN(15), xx, 64, 6, P20, 6, 64, 0.f, 0);        // x branch
    bnstat_k<<<64, 256, 0, stream>>>(xx, P20, 64, mean, rstd);
    bnapply_k<<<cdiv(P20 * 64, 256), 256, 0, stream>>>(xx, P20, 64, IN(16), IN(17), mean, rstd, 0.01f);
    mul_k<<<cdiv(P20 * 64, 256), 256, 0, stream>>>(xx, wB, P20 * 64);
    repack_k<<<cdiv(64L * 64 * 20, 256), 256, 0, stream>>>(IN(18), oWr, 64);
    gemm(stream, oWr, xx, IN(19), x1a, 64, 1280, PTS, 1280, 64, 0.2f, 1);     // o + lrelu(0.2)
    gemm(stream, IN(20), style, IN(21), sbuf, 128, 128, PTS, 128, 128, 0.f, 0); // a1
    instat_k<<<NB * 64, 256, 0, stream>>>(x1a, 64, im, ir);
    adain_k<<<cdiv(PTS * 64, 256), 256, 0, stream>>>(x1a, sbuf, im, ir, 64);

    // ================= Phase C: edge block 2 (C=64, fout=128) ===============
    knn_k<64><<<cdiv(PTS, 256), 256, 0, stream>>>(x1a, idx);
    edge_k<64><<<cdiv(P20, 256), 256, 0, stream>>>(x1a, idx, ebuf);           // (P20,128)
    gemm(stream, IN(22), ebuf + 64, IN(23), wA, 64, 64, P20, 128, 64, 0.f, 0); // w1
    bnstat_k<<<64, 256, 0, stream>>>(wA, P20, 64, mean, rstd);
    bnapply_k<<<cdiv(P20 * 64, 256), 256, 0, stream>>>(wA, P20, 64, IN(24), IN(25), mean, rstd, 0.01f);
    gemm(stream, IN(26), wA, IN(27), wB, 128, 64, P20, 64, 128, 0.f, 0);      // w2
    bnstat_k<<<128, 256, 0, stream>>>(wB, P20, 128, mean, rstd);
    bnapply_k<<<cdiv(P20 * 128, 256), 256, 0, stream>>>(wB, P20, 128, IN(28), IN(29), mean, rstd, 0.01f);
    softmax_k<<<cdiv(PTS * 128, 256), 256, 0, stream>>>(wB, PTS, 128);
    gemm(stream, IN(30), ebuf, IN(31), xx, 128, 128, P20, 128, 128, 0.f, 0);  // x branch
    bnstat_k<<<128, 256, 0, stream>>>(xx, P20, 128, mean, rstd);
    bnapply_k<<<cdiv(P20 * 128, 256), 256, 0, stream>>>(xx, P20, 128, IN(32), IN(33), mean, rstd, 0.01f);
    mul_k<<<cdiv(P20 * 128, 256), 256, 0, stream>>>(xx, wB, P20 * 128);
    repack_k<<<cdiv(128L * 128 * 20, 256), 256, 0, stream>>>(IN(34), oWr, 128);
    gemm(stream, oWr, xx, IN(35), x2a, 128, 2560, PTS, 2560, 128, 0.2f, 1);   // o + lrelu(0.2)
    gemm(stream, IN(36), style, IN(37), sbuf, 256, 128, PTS, 128, 256, 0.f, 0); // a2
    instat_k<<<NB * 128, 256, 0, stream>>>(x2a, 128, im, ir);
    adain_k<<<cdiv(PTS * 128, 256), 256, 0, stream>>>(x2a, sbuf, im, ir, 128);

    // ================= Phase D: global pool + decoder =======================
    colmax_k<<<4, 256, 0, stream>>>(x2a, fg);
    gemm(stream, IN(38), fg, IN(39), fg1, 128, 128, NB, 128, 128, 0.f, 0);    // g1
    bn1_k<<<1, 128, 0, stream>>>(fg1, 128, IN(40), IN(41), 0.01f);
    gemm(stream, IN(42), fg1, IN(43), fg2, 512, 128, NB, 128, 512, 0.f, 0);   // g2
    bn1_k<<<2, 256, 0, stream>>>(fg2, 512, IN(44), IN(45), 0.01f);
    fc_k<<<cdiv(PTS * 640, 256), 256, 0, stream>>>(fg2, x2a, fc);
    gemm(stream, IN(46), fc, IN(47), t1o, 256, 640, PTS, 640, 256, 0.01f, 1); // t1
    gemm(stream, IN(48), t1o, IN(49), t2o, 64, 256, PTS, 256, 64, 0.01f, 1);  // t2
    final_k<<<cdiv(PTS, 256), 256, 0, stream>>>(t2o, IN(50), IN(51), (float*)d_out);
#undef IN
}
